// EdgeMotionModel_9285719294463
// MI455X (gfx1250) — compile-verified, matrix operand order checked by reading, not run on hardware
//
#include <hip/hip_runtime.h>
#include <hip/hip_bf16.h>

// ---------------------------------------------------------------------------
// MI455X (gfx1250): ResNet-18 backbone as implicit-GEMM bf16 WMMA convolutions
// (BN+ReLU+residual fused), LSTM recurrence as per-step WMMA GEMMs.
// wave32; 256-thread blocks (8 waves); each wave owns a 32x32 output tile.
// A tiles staged to LDS via GLOBAL_LOAD_ASYNC_TO_LDS_B128 when available
// (fallback: b128 global load + ds_store). B (weights) pre-swizzled to
// fragment-native layout: a wave's 32x16 fragment is one 32B load per lane.
// ---------------------------------------------------------------------------

typedef __bf16 bf16;
typedef bf16  bf16x8  __attribute__((ext_vector_type(8)));
typedef bf16  bf16x16 __attribute__((ext_vector_type(16)));
typedef float f32x8   __attribute__((ext_vector_type(8)));

#define BB 64   // batch

#ifndef __has_builtin
#define __has_builtin(x) 0
#endif
#if __has_builtin(__builtin_amdgcn_global_load_async_to_lds_b128) && \
    __has_builtin(__builtin_amdgcn_s_wait_asynccnt)
#define ASYNC_LDS 1
typedef int v4i_ __attribute__((ext_vector_type(4)));
typedef __attribute__((address_space(1))) v4i_* as1_v4i;
typedef __attribute__((address_space(3))) v4i_* as3_v4i;
#endif

__device__ __forceinline__ float bf2f(bf16 b) {
    unsigned short h = __builtin_bit_cast(unsigned short, b);
    unsigned u = ((unsigned)h) << 16;
    return __builtin_bit_cast(float, u);
}
__device__ __forceinline__ bf16 f2bf(float f) {
    unsigned u = __builtin_bit_cast(unsigned, f);
    unsigned r = (u + 0x7FFFu + ((u >> 16) & 1u)) >> 16;
    return __builtin_bit_cast(bf16, (unsigned short)r);
}
__device__ __forceinline__ bf16 bfzero() {
    return __builtin_bit_cast(bf16, (unsigned short)0);
}

// 16-bit A 16x32 / B 32x16 fragment k-offset for (lane, j):
// lanes 0-15 hold K {0..7,16..23}, lanes 16-31 hold K {8..15,24..31}
__device__ __forceinline__ int frag_off(int lane, int j) {
    int g = (lane & 16) ? 8 : 0;
    return (j < 8) ? (g + j) : (16 + g + (j - 8));
}

__device__ __forceinline__ f32x8 wmma_bf16(bf16x16 a, bf16x16 b, f32x8 c) {
    return __builtin_amdgcn_wmma_f32_16x16x32_bf16(
        false, a, false, b, (short)0, c, false, false);
}

__device__ __forceinline__ bf16x16 join8(bf16x8 lo, bf16x8 hi) {
    return __builtin_shufflevector(lo, hi, 0, 1, 2, 3, 4, 5, 6, 7,
                                   8, 9, 10, 11, 12, 13, 14, 15);
}

// Stage one 16-byte segment global -> LDS (async path if available).
__device__ __forceinline__ void stage16(const bf16* __restrict__ gp, bf16* lp) {
#ifdef ASYNC_LDS
    __builtin_amdgcn_global_load_async_to_lds_b128((as1_v4i)gp, (as3_v4i)lp, 0, 0);
#else
    *(bf16x8*)lp = *(const bf16x8*)gp;
#endif
}
__device__ __forceinline__ void stage_fence() {
#ifdef ASYNC_LDS
    __builtin_amdgcn_s_wait_asynccnt(0);
#endif
}

// B fragment from pre-swizzled pack: layout [K/32][N/16][lane(32)][16]
// -> one contiguous 32-byte load per lane.
template <int N>
__device__ __forceinline__ bf16x16 load_bpack(const bf16* __restrict__ Wp,
                                              int kb, int nt16, int lane) {
    const size_t idx = (((size_t)(kb >> 5) * (N / 16) + nt16) * 32 + lane);
    return *(const bf16x16*)(Wp + idx * 16);
}

// ---------------------------------------------------------------------------
// Implicit-GEMM convolution: fused BN scale/bias + optional residual + ReLU.
// Block = NG groups x NW waves; group shares one 32-row m-tile (A in LDS),
// each wave in group owns a different 32-col n-tile.
// ---------------------------------------------------------------------------
template <int CIN, int COUT, int KS, int STRIDE, int PAD, int HIN, int WIN,
          bool RELU, bool HASRES>
__global__ void __launch_bounds__(256)
conv_wmma_kernel(const bf16* __restrict__ in, const bf16* __restrict__ wp,
                 const float* __restrict__ sb,   // [0..C)=scale, [C..2C)=bias
                 const bf16* __restrict__ res, bf16* __restrict__ out) {
    constexpr int HOUT = (HIN + 2 * PAD - KS) / STRIDE + 1;
    constexpr int WOUT = (WIN + 2 * PAD - KS) / STRIDE + 1;
    constexpr int KPAD = ((KS * KS * CIN) + 31) & ~31;
    constexpr int TN   = COUT / 32;
    constexpr int NW   = (TN < 8) ? TN : 8;   // waves (n-tiles) per group
    constexpr int NG   = 8 / NW;              // groups (m-tiles) per block
    constexpr int TPG  = NW * 32;             // threads per group
    constexpr int SPT  = (TPG >= 128) ? 1 : (128 / TPG);  // segments per thread
    constexpr int NBN  = TN / NW;             // blocks along N

    __shared__ __align__(16) bf16 lds[NG][32][40];

    const int lane  = threadIdx.x & 31;
    const int widx  = threadIdx.x >> 5;
    const int group = widx / NW;
    const int ntile = widx % NW;
    const int tig   = threadIdx.x & (TPG - 1);
    const int bm    = (int)blockIdx.x / NBN;
    const int bn    = (int)blockIdx.x % NBN;
    const int m0    = (bm * NG + group) * 32;
    const int n0    = (bn * NW + ntile) * 32;

    // ---- k-invariant per-segment staging state ----
    bool seg_act[SPT];
    int  seg_q8[SPT], seg_wo[SPT], seg_ho[SPT];
    const bf16* seg_base[SPT];   // input base for this segment's image
    bf16*       seg_lp[SPT];     // LDS destination
#pragma unroll
    for (int s = 0; s < SPT; ++s) {
        const int seg = tig + s * TPG;    // 128 segments: 32 rows x 4 x 8 elems
        bool act = true;
        if constexpr (TPG > 128) act = (seg < 128);
        seg_act[s] = act;
        const int r  = seg >> 2;
        const int q  = (seg & 3) * 8;
        const int m  = m0 + r;
        const int wo = m % WOUT;
        const int tt = m / WOUT;
        seg_q8[s]   = q;
        seg_wo[s]   = wo;
        seg_ho[s]   = tt % HOUT;
        seg_base[s] = in + ((size_t)(tt / HOUT) * HIN) * WIN * CIN;
        seg_lp[s]   = &lds[group][r][q];
    }

    f32x8 acc00 = {}, acc01 = {}, acc10 = {}, acc11 = {};

    for (int kb = 0; kb < KPAD; kb += 32) {
        // ---- cooperative im2col stage of the group's 32x32 A tile ----
#pragma unroll
        for (int s = 0; s < SPT; ++s) {
            if (seg_act[s]) {
                const int k0 = kb + seg_q8[s];
                if constexpr (CIN >= 8) {
                    const int c0 = k0 % CIN;
                    const int rs = k0 / CIN;
                    const int rr = rs / KS, ss = rs % KS;
                    const int y = seg_ho[s] * STRIDE - PAD + rr;
                    const int x = seg_wo[s] * STRIDE - PAD + ss;
                    if (rs < KS * KS &&
                        (unsigned)y < (unsigned)HIN && (unsigned)x < (unsigned)WIN) {
                        stage16(seg_base[s] + ((size_t)y * WIN + x) * CIN + c0,
                                seg_lp[s]);
                    } else {
                        *(bf16x8*)seg_lp[s] = bf16x8{};
                    }
                } else {
                    bf16x8 d = {};
#pragma unroll
                    for (int e = 0; e < 8; ++e) {
                        const int k  = k0 + e;
                        const int c  = k % CIN;
                        const int rs = k / CIN;
                        bf16 ev = bfzero();
                        if (rs < KS * KS) {
                            const int rr = rs / KS, ss = rs % KS;
                            const int y = seg_ho[s] * STRIDE - PAD + rr;
                            const int x = seg_wo[s] * STRIDE - PAD + ss;
                            if ((unsigned)y < (unsigned)HIN && (unsigned)x < (unsigned)WIN)
                                ev = seg_base[s][((size_t)y * WIN + x) * CIN + c];
                        }
                        d[e] = ev;
                    }
                    *(bf16x8*)seg_lp[s] = d;
                }
            }
        }
        stage_fence();
        __syncthreads();

        // ---- fragments: A from LDS (b128 reads), B from packed global ----
        const int g  = (lane & 16) ? 8 : 0;
        const int lr = lane & 15;
        bf16x16 a0 = join8(*(const bf16x8*)&lds[group][lr][g],
                           *(const bf16x8*)&lds[group][lr][16 + g]);
        bf16x16 a1 = join8(*(const bf16x8*)&lds[group][16 + lr][g],
                           *(const bf16x8*)&lds[group][16 + lr][16 + g]);
        bf16x16 b0 = load_bpack<COUT>(wp, kb, n0 >> 4, lane);
        bf16x16 b1 = load_bpack<COUT>(wp, kb, (n0 >> 4) + 1, lane);
        if (kb + 32 < KPAD)
            __builtin_prefetch(wp + (((size_t)((kb >> 5) + 1) * (COUT / 16) +
                                      (n0 >> 4)) * 32 + lane) * 16, 0, 2);
        acc00 = wmma_bf16(a0, b0, acc00);
        acc01 = wmma_bf16(a0, b1, acc01);
        acc10 = wmma_bf16(a1, b0, acc10);
        acc11 = wmma_bf16(a1, b1, acc11);
        __syncthreads();
    }

    auto st = [&](const f32x8& acc, int mb, int nb2) {
        const int col = nb2 + (lane & 15);
        const int rb  = mb + ((lane & 16) ? 8 : 0);
        const float scl = sb[col];
        const float bia = sb[COUT + col];
#pragma unroll
        for (int r = 0; r < 8; ++r) {
            const size_t row = (size_t)(rb + r);
            float v = acc[r] * scl + bia;
            if constexpr (HASRES) v += bf2f(res[row * COUT + col]);
            if constexpr (RELU) v = fmaxf(v, 0.f);
            out[row * COUT + col] = f2bf(v);
        }
    };
    st(acc00, m0, n0);
    st(acc01, m0, n0 + 16);
    st(acc10, m0 + 16, n0);
    st(acc11, m0 + 16, n0 + 16);
}

// ---------------------------------------------------------------------------
// Generic WMMA GEMM (M = 64): out = act(A1@W1 + [A2@W2] + bias + [Xs@Wsm^T])
// A: [64][K] bf16; W: fragment-packed bf16; Xs:[64][Ks] f32; Wsm:[N][Ks] f32.
// ACT: 0=none, 1=relu, 2=tanh.
// ---------------------------------------------------------------------------
template <int ACT, int N>
__global__ void __launch_bounds__(256)
gemm_wmma_kernel(const bf16* __restrict__ A1, const bf16* __restrict__ W1p, int K1,
                 const bf16* __restrict__ A2, const bf16* __restrict__ W2p, int K2,
                 const float* __restrict__ bias,
                 const float* __restrict__ Xs, const float* __restrict__ Wsm, int Ks,
                 float* __restrict__ outF, bf16* __restrict__ outB) {
    constexpr int TN  = N / 32;
    constexpr int NW  = (TN < 8) ? TN : 8;
    constexpr int NG  = 8 / NW;
    constexpr int TPG = NW * 32;
    constexpr int SPT = (TPG >= 128) ? 1 : (128 / TPG);
    constexpr int NBN = TN / NW;

    __shared__ __align__(16) bf16 lds[NG][32][40];

    const int lane  = threadIdx.x & 31;
    const int widx  = threadIdx.x >> 5;
    const int group = widx / NW;
    const int ntile = widx % NW;
    const int tig   = threadIdx.x & (TPG - 1);
    const int bm    = (int)blockIdx.x / NBN;
    const int bn    = (int)blockIdx.x % NBN;
    const int m0    = (bm * NG + group) * 32;
    const int n0    = (bn * NW + ntile) * 32;

    f32x8 acc00 = {}, acc01 = {}, acc10 = {}, acc11 = {};

    auto kloop = [&](const bf16* __restrict__ A, const bf16* __restrict__ Wp, int K) {
        // k-invariant per-segment staging state
        bool seg_act[SPT];
        const bf16* rowp[SPT];
        bf16* lp[SPT];
#pragma unroll
        for (int s = 0; s < SPT; ++s) {
            const int seg = tig + s * TPG;
            bool act = true;
            if constexpr (TPG > 128) act = (seg < 128);
            seg_act[s] = act;
            const int r = seg >> 2;
            const int q = (seg & 3) * 8;
            rowp[s] = A + (size_t)(m0 + r) * K + q;
            lp[s]   = &lds[group][r][q];
        }
        for (int kb = 0; kb < K; kb += 32) {
#pragma unroll
            for (int s = 0; s < SPT; ++s)
                if (seg_act[s]) stage16(rowp[s] + kb, lp[s]);
            stage_fence();
            __syncthreads();
            const int g  = (lane & 16) ? 8 : 0;
            const int lr = lane & 15;
            bf16x16 a0 = join8(*(const bf16x8*)&lds[group][lr][g],
                               *(const bf16x8*)&lds[group][lr][16 + g]);
            bf16x16 a1 = join8(*(const bf16x8*)&lds[group][16 + lr][g],
                               *(const bf16x8*)&lds[group][16 + lr][16 + g]);
            bf16x16 b0 = load_bpack<N>(Wp, kb, n0 >> 4, lane);
            bf16x16 b1 = load_bpack<N>(Wp, kb, (n0 >> 4) + 1, lane);
            if (kb + 32 < K)
                __builtin_prefetch(Wp + (((size_t)((kb >> 5) + 1) * (N / 16) +
                                          (n0 >> 4)) * 32 + lane) * 16, 0, 2);
            acc00 = wmma_bf16(a0, b0, acc00);
            acc01 = wmma_bf16(a0, b1, acc01);
            acc10 = wmma_bf16(a1, b0, acc10);
            acc11 = wmma_bf16(a1, b1, acc11);
            __syncthreads();
        }
    };
    kloop(A1, W1p, K1);
    if (A2 != nullptr) kloop(A2, W2p, K2);

    auto st = [&](const f32x8& acc, int mb, int nb2) {
        const int col = nb2 + (lane & 15);
        const int rb  = mb + ((lane & 16) ? 8 : 0);
        const float bia = bias ? bias[col] : 0.f;
#pragma unroll
        for (int r = 0; r < 8; ++r) {
            const int row = rb + r;
            float v = acc[r] + bia;
            if (Xs != nullptr) {
                for (int k = 0; k < Ks; ++k)
                    v += Xs[row * Ks + k] * Wsm[col * Ks + k];
            }
            if constexpr (ACT == 1) v = fmaxf(v, 0.f);
            if constexpr (ACT == 2) v = tanhf(v);
            if (outF) outF[(size_t)row * N + col] = v;
            if (outB) outB[(size_t)row * N + col] = f2bf(v);
        }
    };
    st(acc00, m0, n0);
    st(acc01, m0, n0 + 16);
    st(acc10, m0 + 16, n0);
    st(acc11, m0 + 16, n0 + 16);
}

// ---------------------------------------------------------------------------
// Prep / elementwise kernels
// ---------------------------------------------------------------------------

// NCHW f32 (64,3,256,256) -> NHWC bf16 with channels padded to 4 (c3 = 0)
__global__ void prep_img_kernel(const float* __restrict__ img, bf16* __restrict__ out) {
    int idx = blockIdx.x * 256 + threadIdx.x;
    if (idx >= BB * 256 * 256) return;
    const int x = idx & 255;
    const int y = (idx >> 8) & 255;
    const int b = idx >> 16;
    const size_t o   = (size_t)idx * 4;
    const size_t pix = (size_t)y * 256 + x;
    out[o + 0] = f2bf(img[((size_t)b * 3 + 0) * 65536 + pix]);
    out[o + 1] = f2bf(img[((size_t)b * 3 + 1) * 65536 + pix]);
    out[o + 2] = f2bf(img[((size_t)b * 3 + 2) * 65536 + pix]);
    out[o + 3] = bfzero();
}

// OIHW f32 conv weight -> fragment pack [KPAD/32][COUT/16][lane][16] bf16
template <int CINR, int CP, int COUT, int KS>
__global__ void prep_convw_kernel(const float* __restrict__ w, bf16* __restrict__ out) {
    constexpr int KPAD = ((KS * KS * CP) + 31) & ~31;
    int idx = blockIdx.x * 256 + threadIdx.x;
    if (idx >= KPAD * COUT) return;
    const int j    = idx & 15;
    int t          = idx >> 4;
    const int lane = t & 31;
    t >>= 5;
    const int nt = t % (COUT / 16);
    const int kt = t / (COUT / 16);
    const int k  = kt * 32 + frag_off(lane, j);
    const int n  = nt * 16 + (lane & 15);
    const int c  = k % CP;
    const int rs = k / CP;
    float v = 0.f;
    if (rs < KS * KS && c < CINR)
        v = w[((size_t)n * CINR + c) * (KS * KS) + rs];
    out[idx] = f2bf(v);
}

// [N][K] f32 (torch linear weight) -> fragment pack [K/32][N/16][lane][16] bf16
__global__ void prep_wpack_kernel(const float* __restrict__ w, bf16* __restrict__ out,
                                  int N, int K) {
    int idx = blockIdx.x * 256 + threadIdx.x;
    if (idx >= N * K) return;
    const int j    = idx & 15;
    int t          = idx >> 4;
    const int lane = t & 31;
    t >>= 5;
    const int nt = t % (N / 16);
    const int kt = t / (N / 16);
    const int k  = kt * 32 + frag_off(lane, j);
    const int n  = nt * 16 + (lane & 15);
    out[idx] = f2bf(w[(size_t)n * K + k]);
}

// BN inference fold: sb[c] = gamma*rsqrt(var+eps), sb[C+c] = beta - mean*scale
__global__ void prep_bn_kernel(const float* __restrict__ g, const float* __restrict__ b,
                               const float* __restrict__ m, const float* __restrict__ v,
                               float* __restrict__ sb, int C) {
    int i = blockIdx.x * 256 + threadIdx.x;
    if (i >= C) return;
    float inv = g[i] * rsqrtf(v[i] + 1e-5f);
    sb[i]     = inv;
    sb[C + i] = b[i] - m[i] * inv;
}

__global__ void bias_sum_kernel(const float* __restrict__ a, const float* __restrict__ b,
                                float* __restrict__ o, int n) {
    int i = blockIdx.x * 256 + threadIdx.x;
    if (i < n) o[i] = a[i] + b[i];
}

// maxpool 3x3/2 pad1 on NHWC bf16: (64,128,128,64) -> (64,64,64,64)
__global__ void maxpool_kernel(const bf16* __restrict__ in, bf16* __restrict__ out) {
    int idx = blockIdx.x * 256 + threadIdx.x;
    if (idx >= BB * 64 * 64 * 64) return;
    const int c = idx & 63;
    int t = idx >> 6;
    const int x = t & 63; t >>= 6;
    const int y = t & 63;
    const int b = t >> 6;
    float mx = -3.4e38f;
#pragma unroll
    for (int dy = 0; dy < 3; ++dy) {
#pragma unroll
        for (int dx = 0; dx < 3; ++dx) {
            const int iy = 2 * y - 1 + dy;
            const int ix = 2 * x - 1 + dx;
            if ((unsigned)iy < 128u && (unsigned)ix < 128u) {
                float v = bf2f(in[(((size_t)b * 128 + iy) * 128 + ix) * 64 + c]);
                mx = fmaxf(mx, v);
            }
        }
    }
    out[idx] = f2bf(mx);
}

// NHWC (64,8,8,512) -> enc [64][8192] bf16, enc[b][c*16 + y2*4 + x2] = 2x2 mean
__global__ void avgpool_kernel(const bf16* __restrict__ in, bf16* __restrict__ enc) {
    int idx = blockIdx.x * 256 + threadIdx.x;
    if (idx >= BB * 512 * 16) return;
    const int p = idx & 15;
    const int c = (idx >> 4) & 511;
    const int b = idx >> 13;
    const int y2 = p >> 2, x2 = p & 3;
    float s = 0.f;
#pragma unroll
    for (int dy = 0; dy < 2; ++dy)
#pragma unroll
        for (int dx = 0; dx < 2; ++dx)
            s += bf2f(in[(((size_t)b * 8 + 2 * y2 + dy) * 8 + (2 * x2 + dx)) * 512 + c]);
    enc[(size_t)b * 8192 + c * 16 + p] = f2bf(0.25f * s);
}

// h2 = h1, c1 = c2 = 0, pred = 0
__global__ void init_state_kernel(const bf16* __restrict__ h1, bf16* __restrict__ h2,
                                  float* __restrict__ c1, float* __restrict__ c2,
                                  float* __restrict__ pred) {
    int i = blockIdx.x * 256 + threadIdx.x;
    if (i < BB * 256) { h2[i] = h1[i]; c1[i] = 0.f; c2[i] = 0.f; }
    if (i < BB * 5) pred[i] = 0.f;
}

__device__ __forceinline__ float sigmoidf(float x) { return 1.f / (1.f + expf(-x)); }

// gates [64][1024] f32 (i,f,g,o blocks of 256) -> update c (f32) and h (bf16)
__global__ void lstm_cell_kernel(const float* __restrict__ gates, float* __restrict__ c,
                                 bf16* __restrict__ hbf) {
    int t = blockIdx.x * 256 + threadIdx.x;
    if (t >= BB * 256) return;
    const int m = t >> 8;
    const int n = t & 255;
    const float* g = gates + (size_t)m * 1024;
    float i  = sigmoidf(g[n]);
    float f  = sigmoidf(g[256 + n]);
    float gg = tanhf(g[512 + n]);
    float o  = sigmoidf(g[768 + n]);
    float cv = f * c[t] + i * gg;
    c[t] = cv;
    hbf[t] = f2bf(o * tanhf(cv));
}

// pred = hidden(64x128 bf16) @ w2^T(5x128 f32) + b2 ; writes d_out[:,t,:] & predbuf
__global__ void head2_kernel(const bf16* __restrict__ hidden, const float* __restrict__ w2,
                             const float* __restrict__ b2, float* __restrict__ dout,
                             float* __restrict__ predbuf, int t) {
    int i = blockIdx.x * 256 + threadIdx.x;
    if (i >= BB * 5) return;
    const int m = i / 5;
    const int n = i % 5;
    float a = b2[n];
    for (int k = 0; k < 128; ++k)
        a += bf2f(hidden[m * 128 + k]) * w2[n * 128 + k];
    dout[((size_t)m * 256 + t) * 5 + n] = a;
    predbuf[i] = a;
}

// ---------------------------------------------------------------------------
// Host-side helpers
// ---------------------------------------------------------------------------
template <int CINR, int CP, int COUT, int KS, int STRIDE, int PAD, int HIN, int WIN,
          bool RELU, bool HASRES>
static inline void run_conv(hipStream_t stream, const float* wsrc,
                            const float* g, const float* b, const float* m, const float* v,
                            bf16* wbuf, float* sbbuf,
                            const bf16* in, const bf16* res, bf16* out) {
    constexpr int KPAD = ((KS * KS * CP) + 31) & ~31;
    constexpr int HOUT = (HIN + 2 * PAD - KS) / STRIDE + 1;
    constexpr int WOUT = (WIN + 2 * PAD - KS) / STRIDE + 1;
    constexpr int M    = BB * HOUT * WOUT;
    constexpr int TN   = COUT / 32;
    constexpr int NW   = (TN < 8) ? TN : 8;
    constexpr int NG   = 8 / NW;
    constexpr int BLKS = (M / 32 / NG) * (TN / NW);
    prep_convw_kernel<CINR, CP, COUT, KS>
        <<<(KPAD * COUT + 255) / 256, 256, 0, stream>>>(wsrc, wbuf);
    prep_bn_kernel<<<(COUT + 255) / 256, 256, 0, stream>>>(g, b, m, v, sbbuf, COUT);
    conv_wmma_kernel<CP, COUT, KS, STRIDE, PAD, HIN, WIN, RELU, HASRES>
        <<<BLKS, 256, 0, stream>>>(in, wbuf, sbbuf, res, out);
}

extern "C" void kernel_launch(void* const* d_in, const int* in_sizes, int n_in,
                              void* d_out, int out_size, void* d_ws, size_t ws_size,
                              hipStream_t stream) {
    (void)in_sizes; (void)n_in; (void)out_size; (void)ws_size;
    auto F = [&](int i) { return (const float*)d_in[i]; };

    // ---- workspace layout (deterministic bump allocator) ----
    char* base = (char*)d_ws;
    size_t off = 0;
    auto alloc = [&](size_t bytes) -> void* {
        void* r = base + off;
        off = (off + bytes + 255) & ~(size_t)255;
        return r;
    };

    bf16* imgP = (bf16*)alloc((size_t)BB * 256 * 256 * 4 * 2);
    bf16* big0 = (bf16*)alloc((size_t)BB * 128 * 128 * 64 * 2);
    bf16* buf1 = (bf16*)alloc((size_t)BB * 64 * 64 * 64 * 2);
    bf16* buf2 = (bf16*)alloc((size_t)BB * 64 * 64 * 64 * 2);
    bf16* buf3 = (bf16*)alloc((size_t)BB * 64 * 64 * 64 * 2);

    auto wb = [&](int cp, int cout, int ks) {
        return (bf16*)alloc((size_t)(((ks * ks * cp) + 31) & ~31) * cout * 2);
    };
    auto sbuf = [&](int cout) { return (float*)alloc((size_t)cout * 2 * 4); };

    bf16* w_c1 = wb(4, 64, 7);      float* s_c1 = sbuf(64);
    bf16* w_10a = wb(64, 64, 3);    float* s_10a = sbuf(64);
    bf16* w_10b = wb(64, 64, 3);    float* s_10b = sbuf(64);
    bf16* w_11a = wb(64, 64, 3);    float* s_11a = sbuf(64);
    bf16* w_11b = wb(64, 64, 3);    float* s_11b = sbuf(64);
    bf16* w_2d  = wb(64, 128, 1);   float* s_2d  = sbuf(128);
    bf16* w_20a = wb(64, 128, 3);   float* s_20a = sbuf(128);
    bf16* w_20b = wb(128, 128, 3);  float* s_20b = sbuf(128);
    bf16* w_21a = wb(128, 128, 3);  float* s_21a = sbuf(128);
    bf16* w_21b = wb(128, 128, 3);  float* s_21b = sbuf(128);
    bf16* w_3d  = wb(128, 256, 1);  float* s_3d  = sbuf(256);
    bf16* w_30a = wb(128, 256, 3);  float* s_30a = sbuf(256);
    bf16* w_30b = wb(256, 256, 3);  float* s_30b = sbuf(256);
    bf16* w_31a = wb(256, 256, 3);  float* s_31a = sbuf(256);
    bf16* w_31b = wb(256, 256, 3);  float* s_31b = sbuf(256);
    bf16* w_4d  = wb(256, 512, 1);  float* s_4d  = sbuf(512);
    bf16* w_40a = wb(256, 512, 3);  float* s_40a = sbuf(512);
    bf16* w_40b = wb(512, 512, 3);  float* s_40b = sbuf(512);
    bf16* w_41a = wb(512, 512, 3);  float* s_41a = sbuf(512);
    bf16* w_41b = wb(512, 512, 3);  float* s_41b = sbuf(512);

    bf16* fcwP  = (bf16*)alloc((size_t)8192 * 256 * 2);
    bf16* whh0P = (bf16*)alloc((size_t)256 * 1024 * 2);
    bf16* wih1P = (bf16*)alloc((size_t)256 * 1024 * 2);
    bf16* whh1P = (bf16*)alloc((size_t)256 * 1024 * 2);
    bf16* hw1P  = (bf16*)alloc((size_t)256 * 128 * 2);
    bf16* enc   = (bf16*)alloc((size_t)BB * 8192 * 2);
    bf16* h1bf  = (bf16*)alloc((size_t)BB * 256 * 2);
    bf16* h2bf  = (bf16*)alloc((size_t)BB * 256 * 2);
    float* c1   = (float*)alloc((size_t)BB * 256 * 4);
    float* c2   = (float*)alloc((size_t)BB * 256 * 4);
    float* gates = (float*)alloc((size_t)BB * 1024 * 4);
    bf16* hidden = (bf16*)alloc((size_t)BB * 128 * 2);
    float* predb = (float*)alloc((size_t)BB * 5 * 4);
    float* bsum0 = (float*)alloc(1024 * 4);
    float* bsum1 = (float*)alloc(1024 * 4);

    // ---- input prep ----
    prep_img_kernel<<<(BB * 256 * 256 + 255) / 256, 256, 0, stream>>>(F(0), imgP);
    prep_wpack_kernel<<<(256 * 8192 + 255) / 256, 256, 0, stream>>>(F(7), fcwP, 256, 8192);
    prep_wpack_kernel<<<(1024 * 256 + 255) / 256, 256, 0, stream>>>(F(111), whh0P, 1024, 256);
    prep_wpack_kernel<<<(1024 * 256 + 255) / 256, 256, 0, stream>>>(F(114), wih1P, 1024, 256);
    prep_wpack_kernel<<<(1024 * 256 + 255) / 256, 256, 0, stream>>>(F(112), whh1P, 1024, 256);
    prep_wpack_kernel<<<(128 * 256 + 255) / 256, 256, 0, stream>>>(F(10), hw1P, 128, 256);
    bias_sum_kernel<<<4, 256, 0, stream>>>(F(109), F(107), bsum0, 1024);
    bias_sum_kernel<<<4, 256, 0, stream>>>(F(110), F(108), bsum1, 1024);

    // ---- backbone ----
    run_conv<3, 4, 64, 7, 2, 3, 256, 256, true, false>(
        stream, F(5), F(2), F(1), F(3), F(4), w_c1, s_c1, imgP, nullptr, big0);
    maxpool_kernel<<<(BB * 64 * 64 * 64 + 255) / 256, 256, 0, stream>>>(big0, buf1);

    // layer1 (64ch, 64x64), identity shortcuts
    run_conv<64, 64, 64, 3, 1, 1, 64, 64, true, false>(
        stream, F(20), F(13), F(12), F(14), F(15), w_10a, s_10a, buf1, nullptr, buf2);
    run_conv<64, 64, 64, 3, 1, 1, 64, 64, true, true>(
        stream, F(21), F(17), F(16), F(18), F(19), w_10b, s_10b, buf2, buf1, buf1);
    run_conv<64, 64, 64, 3, 1, 1, 64, 64, true, false>(
        stream, F(30), F(23), F(22), F(24), F(25), w_11a, s_11a, buf1, nullptr, buf2);
    run_conv<64, 64, 64, 3, 1, 1, 64, 64, true, true>(
        stream, F(31), F(27), F(26), F(28), F(29), w_11b, s_11b, buf2, buf1, buf1);

    // layer2 (64->128, 64x64 -> 32x32), downsample shortcut
    run_conv<64, 64, 128, 1, 2, 0, 64, 64, false, false>(
        stream, F(46), F(43), F(42), F(44), F(45), w_2d, s_2d, buf1, nullptr, buf3);
    run_conv<64, 64, 128, 3, 2, 1, 64, 64, true, false>(
        stream, F(40), F(33), F(32), F(34), F(35), w_20a, s_20a, buf1, nullptr, buf2);
    run_conv<128, 128, 128, 3, 1, 1, 32, 32, true, true>(
        stream, F(41), F(37), F(36), F(38), F(39), w_20b, s_20b, buf2, buf3, buf3);
    run_conv<128, 128, 128, 3, 1, 1, 32, 32, true, false>(
        stream, F(55), F(48), F(47), F(49), F(50), w_21a, s_21a, buf3, nullptr, buf2);
    run_conv<128, 128, 128, 3, 1, 1, 32, 32, true, true>(
        stream, F(56), F(52), F(51), F(53), F(54), w_21b, s_21b, buf2, buf3, buf3);

    // layer3 (128->256, 32x32 -> 16x16)
    run_conv<128, 128, 256, 1, 2, 0, 32, 32, false, false>(
        stream, F(71), F(68), F(67), F(69), F(70), w_3d, s_3d, buf3, nullptr, buf1);
    run_conv<128, 128, 256, 3, 2, 1, 32, 32, true, false>(
        stream, F(65), F(58), F(57), F(59), F(60), w_30a, s_30a, buf3, nullptr, buf2);
    run_conv<256, 256, 256, 3, 1, 1, 16, 16, true, true>(
        stream, F(66), F(62), F(61), F(63), F(64), w_30b, s_30b, buf2, buf1, buf1);
    run_conv<256, 256, 256, 3, 1, 1, 16, 16, true, false>(
        stream, F(80), F(73), F(72), F(74), F(75), w_31a, s_31a, buf1, nullptr, buf2);
    run_conv<256, 256, 256, 3, 1, 1, 16, 16, true, true>(
        stream, F(81), F(77), F(76), F(78), F(79), w_31b, s_31b, buf2, buf1, buf1);

    // layer4 (256->512, 16x16 -> 8x8)
    run_conv<256, 256, 512, 1, 2, 0, 16, 16, false, false>(
        stream, F(96), F(93), F(92), F(94), F(95), w_4d, s_4d, buf1, nullptr, buf3);
    run_conv<256, 256, 512, 3, 2, 1, 16, 16, true, false>(
        stream, F(90), F(83), F(82), F(84), F(85), w_40a, s_40a, buf1, nullptr, buf2);
    run_conv<512, 512, 512, 3, 1, 1, 8, 8, true, true>(
        stream, F(91), F(87), F(86), F(88), F(89), w_40b, s_40b, buf2, buf3, buf3);
    run_conv<512, 512, 512, 3, 1, 1, 8, 8, true, false>(
        stream, F(105), F(98), F(97), F(99), F(100), w_41a, s_41a, buf3, nullptr, buf2);
    run_conv<512, 512, 512, 3, 1, 1, 8, 8, true, true>(
        stream, F(106), F(102), F(101), F(103), F(104), w_41b, s_41b, buf2, buf3, buf3);

    // ---- encoder head ----
    avgpool_kernel<<<(BB * 512 * 16 + 255) / 256, 256, 0, stream>>>(buf3, enc);
    // h0 = tanh(enc @ fc_w^T + fc_b) : N=256 -> 2 blocks
    gemm_wmma_kernel<2, 256><<<2, 256, 0, stream>>>(
        enc, fcwP, 8192, nullptr, nullptr, 0, F(6),
        nullptr, nullptr, 0, nullptr, h1bf);
    init_state_kernel<<<(BB * 256 + 255) / 256, 256, 0, stream>>>(h1bf, h2bf, c1, c2, predb);

    // ---- LSTM recurrence (graph-captured launch sequence) ----
    float* dout = (float*)d_out;
    for (int t = 0; t < 256; ++t) {
        // gates0 = h1 @ w_hh0^T + (b_ih0 + b_hh0) + x @ w_ih0^T
        gemm_wmma_kernel<0, 1024><<<8, 256, 0, stream>>>(
            h1bf, whh0P, 256, nullptr, nullptr, 0, bsum0,
            predb, F(113), 5, gates, nullptr);
        lstm_cell_kernel<<<64, 256, 0, stream>>>(gates, c1, h1bf);

        // gates1 = h1 @ w_ih1^T + h2 @ w_hh1^T + (b_ih1 + b_hh1)
        gemm_wmma_kernel<0, 1024><<<8, 256, 0, stream>>>(
            h1bf, wih1P, 256, h2bf, whh1P, 256, bsum1,
            nullptr, nullptr, 0, gates, nullptr);
        lstm_cell_kernel<<<64, 256, 0, stream>>>(gates, c2, h2bf);

        // hidden = relu(h2 @ w1^T + b1) : N=128 -> 1 block
        gemm_wmma_kernel<1, 128><<<1, 256, 0, stream>>>(
            h2bf, hw1P, 256, nullptr, nullptr, 0, F(8),
            nullptr, nullptr, 0, nullptr, hidden);

        // pred = hidden @ w2^T + b2 -> d_out[:, t, :] and next-step input
        head2_kernel<<<2, 256, 0, stream>>>(hidden, F(11), F(9), dout, predb, t);
    }
}
